// PatchDivideLayer_42597485642512
// MI455X (gfx1250) — matrix operand order repaired and checked by reading, not compiled, not run on hardware
//
#include <hip/hip_runtime.h>

// ---------------- problem constants (fixed by the reference) ----------------
#define B_      8
#define N_      1024
#define C_      1024     // embed dim == out dim
#define K_      512      // selected patches per batch
#define OUTLEN  (N_ + 3 * K_)   // 2560

typedef __attribute__((ext_vector_type(16))) __bf16 v16bf;
typedef __attribute__((ext_vector_type(8)))  float  v8f;

union Frag {
    uint4    q[2];
    unsigned u[8];
    v16bf    v;
};

// RNE float -> bf16 helpers (pure integer ops, no __bf16 arithmetic)
__device__ __forceinline__ unsigned bf16pack(float a, float b) {
    unsigned ua = __float_as_uint(a);
    unsigned ub = __float_as_uint(b);
    unsigned ra = (ua + 0x7FFFu + ((ua >> 16) & 1u)) >> 16;
    unsigned rb = (ub + 0x7FFFu + ((ub >> 16) & 1u)) >> 16;
    return (ra & 0xFFFFu) | (rb << 16);
}
__device__ __forceinline__ float bf16hi_f(float a) {
    unsigned ua = __float_as_uint(a);
    return __uint_as_float((ua + 0x7FFFu + ((ua >> 16) & 1u)) & 0xFFFF0000u);
}

// ---------------- kernel 1: per-batch mask scan + compaction ----------------
__global__ void scan_kernel(const int* __restrict__ mask,
                            int* __restrict__ starts,
                            int* __restrict__ sel_n,
                            int* __restrict__ sel_start) {
    __shared__ int s[N_];
    const int b = blockIdx.x;
    const int t = threadIdx.x;
    const int m = mask[b * N_ + t];
    s[t] = m;
    __syncthreads();
    #pragma unroll
    for (int off = 1; off < N_; off <<= 1) {
        int v = (t >= off) ? s[t - off] : 0;
        __syncthreads();
        s[t] += v;
        __syncthreads();
    }
    const int incl = s[t];
    const int excl = incl - m;
    const int st   = t + 3 * excl;
    starts[b * N_ + t] = st;
    if (m) {
        sel_n[b * K_ + excl]     = t;
        sel_start[b * K_ + excl] = st;
    }
}

// ---------------- kernel 2: repack W into packed hi/lo B-fragment layout ----
// B fragment as packed here (bf16 32x16, K x N, contiguous k, same as A): lanes 0-15 hold K=0..15 (VGPR i -> K=2i,2i+1),
// lanes 16-31 hold K=16..31.
// Interleaved storage: wpk[frag*512 + lane*16 + i] = hi pair i (i=0..7),
//                      wpk[frag*512 + lane*16 + 8 + i] = lo pair i.
// 64 contiguous bytes per lane -> one 4-wide b128 clause per fragment in GEMM.
// frag index = ((j*32 + kt)*64 + nt),  j=kernel tap, kt=K/32 tile, nt=O/16 tile.
__global__ void repack_w(const float* __restrict__ W,
                         unsigned* __restrict__ wpk) {
    const int gid  = blockIdx.x * blockDim.x + threadIdx.x; // 0..262143
    const int lane = gid & 31;
    const int frag = gid >> 5;           // 0..8191
    const int nt   = frag & 63;
    const int kt   = (frag >> 6) & 31;
    const int j    = frag >> 11;
    const int n    = nt * 16 + (lane & 15);
    const int kb   = kt * 32 + ((lane >> 4) << 4);   // +16 for upper lane half
    const size_t ob = (size_t)frag * 512 + (size_t)lane * 16;
    #pragma unroll
    for (int i = 0; i < 8; ++i) {
        const int k0 = kb + 2 * i;
        const float w0 = W[(size_t)k0 * (C_ * 4) + n * 4 + j];
        const float w1 = W[(size_t)(k0 + 1) * (C_ * 4) + n * 4 + j];
        wpk[ob + i]     = bf16pack(w0, w1);
        wpk[ob + 8 + i] = bf16pack(w0 - bf16hi_f(w0), w1 - bf16hi_f(w1));
    }
}

// ---------------- kernel 3: copy non-selected rows --------------------------
__global__ void copy_unselected(const float* __restrict__ x,
                                const int* __restrict__ mask,
                                const int* __restrict__ starts,
                                float* __restrict__ out) {
    const int bn = blockIdx.x;            // b*N + n
    if (mask[bn]) return;                 // GEMM kernel writes selected rows
    const int b   = bn >> 10;
    const int row = starts[bn];
    const float4* src = (const float4*)(x + (size_t)bn * C_);
    float4* dst = (float4*)(out + ((size_t)b * OUTLEN + row) * C_);
    dst[threadIdx.x] = src[threadIdx.x];
}

// ---------------- kernel 4: bf16x3 WMMA GEMM over selected patches ----------
// Grid: (64 m-blocks of 64 patches, 16 n-blocks of 256 cols of j*O space).
// Block: 256 threads = 8 waves in 2(M) x 4(N); wave tile 32x64 = 2x4 v8f accs.
// Double-buffered LDS A stage: one barrier per K-step; next tile's gathered
// global loads + bf16 split overlap the current step's 24 WMMAs.
__global__ __launch_bounds__(256) void gemm_selected(
    const float* __restrict__ x, const float* __restrict__ bias,
    const int* __restrict__ sel_n, const int* __restrict__ sel_start,
    const unsigned* __restrict__ wpk,
    float* __restrict__ out) {
    // [buf][hi/lo][mfrag 0..3][lane 0..31][8 bf16-pair dwords]  = 16 KB
    __shared__ unsigned ldsA[2][2][4][32][8];

    const int t      = threadIdx.x;
    const int mblock = blockIdx.x;           // 0..63
    const int nblock = blockIdx.y;           // 0..15
    const int batch  = mblock >> 3;          // 512 selected per batch / 64
    const int mbase  = mblock * 64;          // global selected-patch base
    const int j      = nblock >> 2;          // conv kernel tap 0..3

    // --- filler role: thread -> (mfrag, lane, half) ---
    const int f_half = t & 1;                // low/high 4 fragment dwords
    const int f_lane = (t >> 1) & 31;
    const int f_mf   = t >> 6;               // 0..3
    const int f_row  = sel_n[mbase + f_mf * 16 + (f_lane & 15)];
    const float* xrow = x + ((size_t)batch * N_ + f_row) * C_;
    // A k order used here (contiguous, same as B packing): lanes0-15 K=0..15, lanes16-31 K=16..31
    const int f_koff = (f_half << 3) + ((f_lane >> 4) << 4);

    // --- compute role ---
    const int lane = t & 31;
    const int w    = t >> 5;
    const int mw   = w >> 2;                 // 0..1
    const int nw   = w & 3;                  // 0..3

    v8f acc[2][4] = {};

    auto stage = [&](int kt, int buf) {
        const float4* src = (const float4*)(xrow + kt * 32 + f_koff);
        const float4 fa = src[0];
        const float4 fb = src[1];
        uint4 hq, lq;
        hq.x = bf16pack(fa.x, fa.y);
        hq.y = bf16pack(fa.z, fa.w);
        hq.z = bf16pack(fb.x, fb.y);
        hq.w = bf16pack(fb.z, fb.w);
        lq.x = bf16pack(fa.x - bf16hi_f(fa.x), fa.y - bf16hi_f(fa.y));
        lq.y = bf16pack(fa.z - bf16hi_f(fa.z), fa.w - bf16hi_f(fa.w));
        lq.z = bf16pack(fb.x - bf16hi_f(fb.x), fb.y - bf16hi_f(fb.y));
        lq.w = bf16pack(fb.z - bf16hi_f(fb.z), fb.w - bf16hi_f(fb.w));
        *(uint4*)&ldsA[buf][0][f_mf][f_lane][f_half * 4] = hq;
        *(uint4*)&ldsA[buf][1][f_mf][f_lane][f_half * 4] = lq;
    };

    stage(0, 0);

    #pragma unroll 2
    for (int kt = 0; kt < 32; ++kt) {
        const int buf = kt & 1;
        __syncthreads();   // stage(kt) complete; also guards buf reuse

        // ---- pull this wave's A fragments into registers ----
        Frag ah[2], al[2];
        #pragma unroll
        for (int fm = 0; fm < 2; ++fm) {
            const int mf = mw * 2 + fm;
            ah[fm].q[0] = *(const uint4*)&ldsA[buf][0][mf][lane][0];
            ah[fm].q[1] = *(const uint4*)&ldsA[buf][0][mf][lane][4];
            al[fm].q[0] = *(const uint4*)&ldsA[buf][1][mf][lane][0];
            al[fm].q[1] = *(const uint4*)&ldsA[buf][1][mf][lane][4];
        }

        // ---- prefetch + stage next A tile into the other buffer ----
        if (kt + 1 < 32) stage(kt + 1, buf ^ 1);

        // ---- B fragments (hi/lo interleaved) + 3-term WMMA ----
        #pragma unroll
        for (int fn = 0; fn < 4; ++fn) {
            const int nt = (nblock & 3) * 16 + nw * 4 + fn;
            const uint4* p = (const uint4*)(wpk
                + ((size_t)((j * 32 + kt) * 64 + nt)) * 512 + (size_t)lane * 16);
            Frag bh, bl;
            bh.q[0] = p[0];
            bh.q[1] = p[1];
            bl.q[0] = p[2];
            bl.q[1] = p[3];
            #pragma unroll
            for (int fm = 0; fm < 2; ++fm) {
                acc[fm][fn] = __builtin_amdgcn_wmma_f32_16x16x32_bf16(
                    false, ah[fm].v, false, bh.v, (short)0, acc[fm][fn], false, false);
                acc[fm][fn] = __builtin_amdgcn_wmma_f32_16x16x32_bf16(
                    false, ah[fm].v, false, bl.v, (short)0, acc[fm][fn], false, false);
                acc[fm][fn] = __builtin_amdgcn_wmma_f32_16x16x32_bf16(
                    false, al[fm].v, false, bh.v, (short)0, acc[fm][fn], false, false);
            }
        }
    }

    // ---- epilogue: bias add + scatter to output rows ----
    #pragma unroll
    for (int fm = 0; fm < 2; ++fm) {
        // C/D layout: lanes0-15 -> M=v, lanes16-31 -> M=v+8; N = lane%16
        const int gmb = mbase + mw * 32 + fm * 16 + ((lane >> 4) << 3);
        int srow[8];
        #pragma unroll
        for (int v = 0; v < 8; ++v) srow[v] = sel_start[gmb + v];
        #pragma unroll
        for (int fn = 0; fn < 4; ++fn) {
            const int nt = (nblock & 3) * 16 + nw * 4 + fn;
            const int o  = nt * 16 + (lane & 15);
            const float bv = bias[o];
            #pragma unroll
            for (int v = 0; v < 8; ++v) {
                out[((size_t)batch * OUTLEN + srow[v] + j) * C_ + o] = acc[fm][fn][v] + bv;
            }
        }
    }
}

// ---------------- host-side launch ------------------------------------------
extern "C" void kernel_launch(void* const* d_in, const int* in_sizes, int n_in,
                              void* d_out, int out_size, void* d_ws, size_t ws_size,
                              hipStream_t stream) {
    const float* x    = (const float*)d_in[0];
    const float* W    = (const float*)d_in[1];
    const float* bias = (const float*)d_in[2];
    const int*   mask = (const int*)d_in[3];
    float* out = (float*)d_out;

    char* ws = (char*)d_ws;
    int* starts    = (int*)(ws);                       // 8192 ints
    int* sel_n     = (int*)(ws + 32768);               // 4096 ints
    int* sel_start = (int*)(ws + 49152);               // 4096 ints
    unsigned* wpk  = (unsigned*)(ws + 65536);          // 16 MB packed hi/lo

    scan_kernel<<<B_, N_, 0, stream>>>(mask, starts, sel_n, sel_start);
    repack_w<<<1024, 256, 0, stream>>>(W, wpk);
    copy_unselected<<<B_ * N_, 256, 0, stream>>>(x, mask, starts, out);
    gemm_selected<<<dim3(64, 16), 256, 0, stream>>>(x, bias, sel_n, sel_start,
                                                    wpk, out);
}
